// Decoder_55362128445508
// MI455X (gfx1250) — compile-verified, matrix operand order checked by reading, not run on hardware
//
#include <hip/hip_runtime.h>
#include <hip/hip_bf16.h>
#include <math.h>
#include <stdint.h>

// Problem constants (from reference)
#define VV 32000
#define EE 512
#define HH 1024
#define LL 2
#define AA 1024
#define SS 1024
#define BB 64

typedef __attribute__((ext_vector_type(2)))  float  v2f;
typedef __attribute__((ext_vector_type(4)))  float  v4f;
typedef __attribute__((ext_vector_type(8)))  float  v8f;
typedef __attribute__((ext_vector_type(4)))  __bf16 v4bf;
typedef __attribute__((ext_vector_type(8)))  __bf16 v8bf;
typedef __attribute__((ext_vector_type(16))) __bf16 v16bf;

// ---------------------------------------------------------------------------
// Generic fp32 WMMA GEMM: C[m0:m0+16, n0:n0+16] = A(MxK) · B(NxK)^T + bias
// One wave per 16x16 output tile, K streamed in steps of 4 via
// v_wmma_f32_16x16x4_f32.  A-frag: lane L holds M=L&15, K=2*(L>>4)+{0,1}.
// B-frag mirrors with N=L&15.  C/D: lane L holds N=L&15, M=v+8*(L>>4).
// ---------------------------------------------------------------------------
__global__ __launch_bounds__(32)
void wmma_gemm_f32(const float* __restrict__ A, int lda,
                   const float* __restrict__ Bw, int ldb, int bcol,
                   const float* __restrict__ bias,
                   float* __restrict__ C, int ldc, int K) {
  const int lane = threadIdx.x;
  const int n0 = blockIdx.x * 16;
  const int m0 = blockIdx.y * 16;
  const int nn = lane & 15;
  const int hi = lane >> 4;

  const float* __restrict__ arow = A  + (size_t)(m0 + nn) * lda;
  const float* __restrict__ brow = Bw + (size_t)(n0 + nn) * ldb + bcol;

  v8f acc = {};
  for (int k = 0; k < K; k += 64) {
    // stream-ahead hints, WGP scope (pull into all cache levels)
    __builtin_prefetch(brow + k + 64, 0, 3);
    __builtin_prefetch(arow + k + 64, 0, 3);
#pragma unroll
    for (int kk = 0; kk < 64; kk += 4) {
      v2f af = *(const v2f*)(arow + k + kk + 2 * hi);
      v2f bf = *(const v2f*)(brow + k + kk + 2 * hi);
      acc = __builtin_amdgcn_wmma_f32_16x16x4_f32(false, af, false, bf,
                                                  (short)0, acc, false, false);
    }
  }

  float bv = bias ? bias[n0 + nn] : 0.0f;
#pragma unroll
  for (int v = 0; v < 8; ++v) {
    int m = v + 8 * hi;
    C[(size_t)(m0 + m) * ldc + (n0 + nn)] = acc[v] + bv;
  }
}

// ---------------------------------------------------------------------------
// Convert W_enc = attn_w1[:, 0:H] to bf16 (stays L2-resident, 2 MB)
// ---------------------------------------------------------------------------
__global__ __launch_bounds__(256)
void cvt_wenc_bf16(const float* __restrict__ w1, __bf16* __restrict__ wb) {
  int i = blockIdx.x * 256 + threadIdx.x;   // over A*H
  int a = i >> 10, h = i & (HH - 1);
  wb[i] = (__bf16)w1[(size_t)a * (2 * HH) + h];
}

// ---------------------------------------------------------------------------
// Fused attention scores:
//   e[r] = sum_a w2[a] * tanh( enc[r,:]·W_enc[a,:] + dproj[b(r),a] )
// r = s*B+b flattened (65536 rows).  One wave per 16-row tile.
// enc tile staged once in LDS as bf16 (padded stride = 1032 elems so row
// stride is 516 dwords -> 4-bank rotation, conflict-free ds_load_b128).
// w2 staged via CDNA5 async global->LDS DMA (ASYNCcnt path).
// K streamed with v_wmma_f32_16x16x32_bf16, f32 accumulate.
// tanh/softmax/context smooth bf16 operand rounding; deterministic (no atomics).
// ---------------------------------------------------------------------------
#define TSTR 1032
__global__ __launch_bounds__(32)
void attn_scores(const float*  __restrict__ enc,    // (S*B, H) f32
                 const __bf16* __restrict__ wencb,  // (A, H) bf16
                 const float*  __restrict__ dproj,  // (B, A) f32
                 const float*  __restrict__ w2,     // (A,) f32
                 float*        __restrict__ e)      // (S*B,) f32
{
  __shared__ __bf16 atile[16 * TSTR];
  __shared__ float  w2s[AA];
  const int lane = threadIdx.x;
  const int r0 = blockIdx.x * 16;
  const int b0 = r0 & (BB - 1);   // 16 consecutive rows stay in one s

  // ---- async DMA: stage w2 (4KB) into LDS, 512B per issue ----------------
#pragma unroll
  for (int ch = 0; ch < 8; ++ch) {
    int idx = (ch * 32 + lane) * 4;                 // float4 per lane
    unsigned lds_addr = (unsigned)(uintptr_t)(&w2s[idx]);
    const float* gsrc = w2 + idx;
    asm volatile("global_load_async_to_lds_b128 %0, %1, off"
                 :: "v"(lds_addr), "v"(gsrc) : "memory");
  }

  // ---- stage 16 rows of enc into LDS as bf16 (coalesced float4 stream) ----
  for (int row = 0; row < 16; ++row) {
    const float* __restrict__ src = enc + (size_t)(r0 + row) * HH;
    __bf16* __restrict__ dst = atile + row * TSTR;
#pragma unroll
    for (int ch = 0; ch < 8; ++ch) {
      int col = (ch * 32 + lane) * 4;
      v4f f = *(const v4f*)(src + col);
      v4bf p;
      p[0] = (__bf16)f[0];
      p[1] = (__bf16)f[1];
      p[2] = (__bf16)f[2];
      p[3] = (__bf16)f[3];
      *(v4bf*)(dst + col) = p;                      // one ds_store_b64
    }
  }
  asm volatile("s_wait_asynccnt 0" ::: "memory");
  __syncthreads();

  const int nn = lane & 15;
  const int hi = lane >> 4;
  float es[8] = {0.f, 0.f, 0.f, 0.f, 0.f, 0.f, 0.f, 0.f};

  for (int a0 = 0; a0 < AA; a0 += 16) {
    v8f acc = {};
    const __bf16* __restrict__ brow = wencb + (size_t)(a0 + nn) * HH;
    const __bf16* __restrict__ arow = atile + nn * TSTR;
    for (int k0 = 0; k0 < HH; k0 += 32) {
      // B fragment: lane holds N=nn, K = k0 + 8*hi + {0..7} and +16
      v8bf blo = *(const v8bf*)(brow + k0 + 8 * hi);
      v8bf bhi = *(const v8bf*)(brow + k0 + 16 + 8 * hi);
      v16bf bf = __builtin_shufflevector(blo, bhi, 0, 1, 2, 3, 4, 5, 6, 7,
                                         8, 9, 10, 11, 12, 13, 14, 15);
      // A fragment from LDS: lane holds M=nn, same K pattern
      v8bf alo = *(const v8bf*)(arow + k0 + 8 * hi);
      v8bf ahi = *(const v8bf*)(arow + k0 + 16 + 8 * hi);
      v16bf af = __builtin_shufflevector(alo, ahi, 0, 1, 2, 3, 4, 5, 6, 7,
                                         8, 9, 10, 11, 12, 13, 14, 15);
      acc = __builtin_amdgcn_wmma_f32_16x16x32_bf16(false, af, false, bf,
                                                    (short)0, acc, false, false);
    }
    float w2v = w2s[a0 + nn];
#pragma unroll
    for (int v = 0; v < 8; ++v) {
      int m = v + 8 * hi;
      float f = acc[v] + dproj[(size_t)(b0 + m) * AA + (a0 + nn)];
      es[v] += w2v * tanhf(f);
    }
  }

  // reduce each row's partial over the 16 N-lanes, write e[r]
#pragma unroll
  for (int v = 0; v < 8; ++v) {
    float s = es[v];
    s += __shfl_xor(s, 8, 16);
    s += __shfl_xor(s, 4, 16);
    s += __shfl_xor(s, 2, 16);
    s += __shfl_xor(s, 1, 16);
    if (nn == 0) e[r0 + v + 8 * hi] = s;
  }
}

// ---------------------------------------------------------------------------
// Per-batch softmax over S + context c[b,h] = sum_s alpha[s]*enc[s,b,h]
// ---------------------------------------------------------------------------
__global__ __launch_bounds__(256)
void softmax_ctx(const float* __restrict__ e,    // (S*B,)
                 const float* __restrict__ enc,  // (S,B,H)
                 float* __restrict__ ctx)        // (B,H)
{
  __shared__ float red[256];
  __shared__ float alpha[SS];
  const int b = blockIdx.x;
  const int t = threadIdx.x;

  float mx = -3.0e38f;
  for (int s = t; s < SS; s += 256) mx = fmaxf(mx, e[s * BB + b]);
  red[t] = mx; __syncthreads();
  for (int w = 128; w >= 1; w >>= 1) {
    if (t < w) red[t] = fmaxf(red[t], red[t + w]);
    __syncthreads();
  }
  mx = red[0]; __syncthreads();

  float sm = 0.f;
  for (int s = t; s < SS; s += 256) {
    float v = expf(e[s * BB + b] - mx);
    alpha[s] = v;
    sm += v;
  }
  red[t] = sm; __syncthreads();
  for (int w = 128; w >= 1; w >>= 1) {
    if (t < w) red[t] += red[t + w];
    __syncthreads();
  }
  const float inv = 1.0f / red[0];

  const int h0 = t * 4;      // 256 threads x 4 = H
  v4f acc = {};
  for (int s = 0; s < SS; ++s) {
    float a = alpha[s] * inv;
    v4f v = *(const v4f*)(enc + ((size_t)s * BB + b) * HH + h0);
    acc += a * v;
  }
  *(v4f*)(ctx + (size_t)b * HH + h0) = acc;
}

// ---------------------------------------------------------------------------
// x = [ embedding[cur_input[b]] , context[b] ]   (B, E+H)
// ---------------------------------------------------------------------------
__global__ __launch_bounds__(256)
void build_x(const int* __restrict__ cur, const float* __restrict__ emb,
             const float* __restrict__ ctx, float* __restrict__ x) {
  int i = blockIdx.x * 256 + threadIdx.x;   // over B*(E+H)
  int b = i / (EE + HH);
  int col = i - b * (EE + HH);
  x[i] = (col < EE) ? emb[(size_t)cur[b] * EE + col]
                    : ctx[(size_t)b * HH + (col - EE)];
}

// ---------------------------------------------------------------------------
// GRU gate math (gi/gh already include biases from the GEMMs)
// ---------------------------------------------------------------------------
__global__ __launch_bounds__(256)
void gru_gate(const float* __restrict__ gi, const float* __restrict__ gh,
              const float* __restrict__ hprev, float* __restrict__ hout) {
  int i = blockIdx.x * 256 + threadIdx.x;   // over B*H
  int b = i >> 10;
  int j = i & (HH - 1);
  const float* __restrict__ gib = gi + (size_t)b * (3 * HH);
  const float* __restrict__ ghb = gh + (size_t)b * (3 * HH);
  float r = 1.0f / (1.0f + expf(-(gib[j] + ghb[j])));
  float z = 1.0f / (1.0f + expf(-(gib[HH + j] + ghb[HH + j])));
  float n = tanhf(gib[2 * HH + j] + r * ghb[2 * HH + j]);
  hout[i] = (1.0f - z) * n + z * hprev[i];
}

// ---------------------------------------------------------------------------
extern "C" void kernel_launch(void* const* d_in, const int* in_sizes, int n_in,
                              void* d_out, int out_size, void* d_ws, size_t ws_size,
                              hipStream_t stream) {
  const int*   cur   = (const int*)  d_in[0];
  const float* state = (const float*)d_in[1];   // (L,B,H)
  const float* enc   = (const float*)d_in[2];   // (S,B,H)
  const float* emb   = (const float*)d_in[3];
  const float* aw1   = (const float*)d_in[4];   // (A,2H)
  const float* aw2   = (const float*)d_in[5];   // (1,A)
  const float* wih0  = (const float*)d_in[6];
  const float* whh0  = (const float*)d_in[7];
  const float* bih0  = (const float*)d_in[8];
  const float* bhh0  = (const float*)d_in[9];
  const float* wih1  = (const float*)d_in[10];
  const float* whh1  = (const float*)d_in[11];
  const float* bih1  = (const float*)d_in[12];
  const float* bhh1  = (const float*)d_in[13];
  const float* outw  = (const float*)d_in[14];
  const float* outb  = (const float*)d_in[15];

  float* out = (float*)d_out;             // [B*V logits][B*H h0][B*H h1]
  float* h0  = out + (size_t)BB * VV;
  float* h1  = h0 + (size_t)BB * HH;

  // workspace layout (floats)
  float* ws    = (float*)d_ws;
  float* dproj = ws;                           // B*A      = 65536
  float* e     = dproj + (size_t)BB * AA;      // S*B      = 65536
  float* ctx   = e     + (size_t)SS * BB;      // B*H      = 65536
  float* x     = ctx   + (size_t)BB * HH;      // B*(E+H)  = 98304
  float* gi    = x     + (size_t)BB * (EE + HH);     // B*3H = 196608
  float* gh    = gi    + (size_t)BB * 3 * HH;        // B*3H = 196608
  __bf16* wencb = (__bf16*)(gh + (size_t)BB * 3 * HH); // A*H bf16 = 2MB

  const float* dec    = state + (size_t)(LL - 1) * BB * HH;  // state[-1]
  const float* state0 = state;
  const float* state1 = state + (size_t)BB * HH;

  // 1) W_enc -> bf16 (L2 resident)
  cvt_wenc_bf16<<<(AA * HH) / 256, 256, 0, stream>>>(aw1, wencb);

  // 2) dproj = dec · W_dec^T   (fp32 WMMA; W_dec = attn_w1[:, H:2H])
  wmma_gemm_f32<<<dim3(AA / 16, BB / 16), 32, 0, stream>>>(
      dec, HH, aw1, 2 * HH, HH, nullptr, dproj, AA, HH);

  // 3) fused attention scores (bf16 WMMA + async LDS staging)
  attn_scores<<<(SS * BB) / 16, 32, 0, stream>>>(enc, wencb, dproj, aw2, e);

  // 4) softmax over S + context
  softmax_ctx<<<BB, 256, 0, stream>>>(e, enc, ctx);

  // 5) x = [embedding gather, context]
  build_x<<<(BB * (EE + HH)) / 256, 256, 0, stream>>>(cur, emb, ctx, x);

  // 6) GRU layer 0
  wmma_gemm_f32<<<dim3(3 * HH / 16, BB / 16), 32, 0, stream>>>(
      x, EE + HH, wih0, EE + HH, 0, bih0, gi, 3 * HH, EE + HH);
  wmma_gemm_f32<<<dim3(3 * HH / 16, BB / 16), 32, 0, stream>>>(
      state0, HH, whh0, HH, 0, bhh0, gh, 3 * HH, HH);
  gru_gate<<<(BB * HH) / 256, 256, 0, stream>>>(gi, gh, state0, h0);

  // 7) GRU layer 1
  wmma_gemm_f32<<<dim3(3 * HH / 16, BB / 16), 32, 0, stream>>>(
      h0, HH, wih1, HH, 0, bih1, gi, 3 * HH, HH);
  wmma_gemm_f32<<<dim3(3 * HH / 16, BB / 16), 32, 0, stream>>>(
      state1, HH, whh1, HH, 0, bhh1, gh, 3 * HH, HH);
  gru_gate<<<(BB * HH) / 256, 256, 0, stream>>>(gi, gh, state1, h1);

  // 8) logits = h1 · out_w^T + out_b   (fp32 WMMA, out_w L2-resident)
  wmma_gemm_f32<<<dim3(VV / 16, BB / 16), 32, 0, stream>>>(
      h1, HH, outw, HH, 0, outb, out, VV, HH);
}
